// RNN_71124658421760
// MI455X (gfx1250) — 8-bit matrix operand order repaired and checked by reading, compiled, not run on hardware
//
#include <hip/hip_runtime.h>

typedef __attribute__((ext_vector_type(16))) int   v16i;
typedef __attribute__((ext_vector_type(8)))  float v8f;

#define DH 512
#define BB 64
#define TT 1024

__device__ __forceinline__ float sigmoidf(float x) {
  return __builtin_amdgcn_rcpf(1.0f + __expf(-x));
}

__device__ __forceinline__ unsigned fp8e4m3(float f) {
  unsigned u = __float_as_uint(f);
  unsigned r = u + 0x7FFFFu + ((u >> 20) & 1u);
  int e = (int)(r >> 23) - 127;
  if (e < -6) {
    int mi = __float2int_rn(f * 512.0f);
    return (unsigned)(mi > 7 ? 7 : mi);
  }
  return (unsigned)(((e + 7) << 3) | ((r >> 20) & 7u));
}
__device__ __forceinline__ unsigned pack4fp8(float a, float b, float c, float d) {
  return fp8e4m3(a) | (fp8e4m3(b) << 8) | (fp8e4m3(c) << 16) | (fp8e4m3(d) << 24);
}

__global__ void RNN_wprep(const float* __restrict__ wraw,
                          const float* __restrict__ m,
                          float* __restrict__ w) {
  int i = blockIdx.x * blockDim.x + threadIdx.x;
  int h = i >> 9, d = i & (DH - 1);
  float acc = 0.0f;
  for (int k = 0; k < DH; ++k)
    acc = fmaf(wraw[h * DH + k], m[k * DH + d], acc);
  w[i] = fabsf(acc);
}

__global__ __launch_bounds__(512, 1) void RNN_recur(
    const float* __restrict__ u, const float* __restrict__ r0,
    const float* __restrict__ noise, const float* __restrict__ win,
    const float* __restrict__ taus, const float* __restrict__ w,
    float* __restrict__ xs) {
  __shared__ unsigned char rT[2][16][DH];

  const int lane  = threadIdx.x & 31;
  const int wv    = threadIdx.x >> 5;
  const int half  = lane >> 4;
  const int nloc  = lane & 15;
  const int off8  = half << 3;
  const int b     = (blockIdx.x << 4) + nloc;

  int mbase[2] = { wv << 5, (wv << 5) + 16 };

  v16i A[2][4];
#pragma unroll
  for (int j = 0; j < 2; ++j) {
    const int row = mbase[j] + nloc;
#pragma unroll
    for (int c = 0; c < 4; ++c) {
      union { v16i v; unsigned q[16]; } au;
#pragma unroll
      for (int v = 0; v < 16; ++v) {
        int ks = (c << 7) + ((v >> 2) << 5) + (half << 4)
               + ((v & 3) << 2);
        const float4 wf = *(const float4*)&w[row * DH + ks];
        au.q[v] = pack4fp8(wf.x, wf.y, wf.z, wf.w);
      }
      A[j][c] = au.v;
    }
  }

  float alp[2][8], wnv[2][8], carry[2][8];
#pragma unroll
  for (int j = 0; j < 2; ++j)
#pragma unroll
    for (int v = 0; v < 8; ++v) {
      int h = mbase[j] + off8 + v;
      alp[j][v]   = 1.0f / taus[h];
      wnv[j][v]   = win[h];
      carry[j][v] = r0[b * DH + h];
    }

  for (int t = 0; t < TT; ++t) {
    const int buf = t & 1;
#pragma unroll
    for (int j = 0; j < 2; ++j) {
      float r[8];
#pragma unroll
      for (int v = 0; v < 8; ++v) r[v] = sigmoidf(carry[j][v]);
      uint2 pk;
      pk.x = pack4fp8(r[0], r[1], r[2], r[3]);
      pk.y = pack4fp8(r[4], r[5], r[6], r[7]);
      *(uint2*)&rT[buf][nloc][mbase[j] + off8] = pk;
    }

    float uval = u[b * TT + t];
    const float* np = noise + ((size_t)t * BB * DH + b * DH);
    float4 nz[2][2];
#pragma unroll
    for (int j = 0; j < 2; ++j) {
      nz[j][0] = *(const float4*)(np + mbase[j] + off8);
      nz[j][1] = *(const float4*)(np + mbase[j] + off8 + 4);
    }

    __syncthreads();

    union Bu { v16i v; uint4 q[4]; } bu[4];
#pragma unroll
    for (int c = 0; c < 4; ++c)
#pragma unroll
      for (int g = 0; g < 4; ++g)
        bu[c].q[g] = *(const uint4*)&rT[buf][nloc][(c << 7) + (g << 5) + (half << 4)];

    v8f acc0 = {}, acc1 = {};
#pragma unroll
    for (int c = 0; c < 4; ++c) {
      acc0 = __builtin_amdgcn_wmma_f32_16x16x128_fp8_fp8(
          A[0][c], bu[c].v, (short)0, acc0, false, false);
      acc1 = __builtin_amdgcn_wmma_f32_16x16x128_fp8_fp8(
          A[1][c], bu[c].v, (short)0, acc1, false, false);
    }

#pragma unroll
    for (int j = 0; j < 2; ++j) {
      float accv[8] = { (j ? acc1 : acc0)[0], (j ? acc1 : acc0)[1],
                        (j ? acc1 : acc0)[2], (j ? acc1 : acc0)[3],
                        (j ? acc1 : acc0)[4], (j ? acc1 : acc0)[5],
                        (j ? acc1 : acc0)[6], (j ? acc1 : acc0)[7] };
      float nzv[8] = { nz[j][0].x, nz[j][0].y, nz[j][0].z, nz[j][0].w,
                       nz[j][1].x, nz[j][1].y, nz[j][1].z, nz[j][1].w };
#pragma unroll
      for (int v = 0; v < 8; ++v) {
        int h = mbase[j] + off8 + v;
        float xnv = (1.0f - alp[j][v]) * carry[j][v] + alp[j][v] * accv[v]
                  + wnv[j][v] * uval + 0.1f * nzv[v];
        carry[j][v] = sigmoidf(xnv);
        xs[(size_t)(b * DH + h) * TT + t] = xnv;
      }
    }
  }
}

__global__ void RNN_out(const float* __restrict__ wout,
                        const float* __restrict__ bias,
                        const float* __restrict__ xs,
                        float* __restrict__ outp) {
  int i = blockIdx.x * blockDim.x + threadIdx.x;
  int b = i >> 10, t = i & (TT - 1);
  float acc = 0.0f;
  for (int h = 0; h < DH; ++h)
    acc = fmaf(wout[h], sigmoidf(xs[(size_t)(b * DH + h) * TT + t]), acc);
  outp[i] = acc + bias[0];
}

extern "C" void kernel_launch(void* const* d_in, const int* in_sizes, int n_in,
                              void* d_out, int out_size, void* d_ws, size_t ws_size,
                              hipStream_t stream) {
  const float* u     = (const float*)d_in[0];
  const float* r0    = (const float*)d_in[1];
  const float* noise = (const float*)d_in[2];
  const float* win   = (const float*)d_in[3];
  const float* m     = (const float*)d_in[4];
  const float* wout  = (const float*)d_in[5];
  const float* wraw  = (const float*)d_in[6];
  const float* taus  = (const float*)d_in[7];
  const float* bias  = (const float*)d_in[8];

  float* outp = (float*)d_out;
  float* xs   = outp + BB * TT;
  float* w    = (float*)d_ws;

  RNN_wprep<<<DH * DH / 256, 256, 0, stream>>>(wraw, m, w);
  RNN_recur<<<4, 512, 0, stream>>>(u, r0, noise, win, taus, w, xs);
  RNN_out<<<BB * TT / 256, 256, 0, stream>>>(wout, bias, xs, outp);
}
